// MultiPoleScoreNet_18382460027271
// MI455X (gfx1250) — matrix operand order repaired and checked by reading, not compiled, not run on hardware
//
#include <hip/hip_runtime.h>
#include <hip/hip_bf16.h>
#include <stdint.h>

typedef __attribute__((ext_vector_type(16))) _Float16 v16h;
typedef __attribute__((ext_vector_type(8)))  _Float16 v8h;
typedef __attribute__((ext_vector_type(8)))  float    v8f;

#define N_PTS 16384
#define KNB   128
#define HID   64
#define H1S   72   // padded row stride (halves) for lds_h1
#define W2S   72   // padded col stride (halves) for lds_w2t

// jax.nn.gelu (approximate=True): 0.5*x*(1+tanh(sqrt(2/pi)*(x+0.044715 x^3)))
// Branch-free tanh: tanh(u) = 1 - 2/(exp(2u)+1), exp via v_exp_f32 (exp2),
// reciprocal via v_rcp_f32. Saturates cleanly to +/-1 for large |u| (no NaN),
// and keeps EXEC untouched (WMMA requires EXEC all-ones in this region).
__device__ __forceinline__ float gelu_tanh(float x) {
    float u = 0.7978845608028654f * (x + 0.044715f * x * x * x);
    float e = __builtin_amdgcn_exp2f(2.8853900817779268f * u);   // exp(2u)
    float t = 1.0f - 2.0f * __builtin_amdgcn_rcpf(e + 1.0f);
    return 0.5f * x * (1.0f + t);
}

__launch_bounds__(256, 2)
__global__ void mpscore_kernel(const float* __restrict__ coords,   // (2, N)
                               const float* __restrict__ samples,  // (8, N)
                               const float* __restrict__ W1,       // (4, 64)
                               const float* __restrict__ b1,       // (64)
                               const float* __restrict__ W2,       // (64, 64)
                               const float* __restrict__ b2,       // (64)
                               const float* __restrict__ W3,       // (64, 1)
                               const float* __restrict__ b3,       // (1)
                               const int*   __restrict__ nbr_idx,  // (N, 128)
                               const uint8_t* __restrict__ nbr_msk,// (N, 128) bool
                               float* __restrict__ out)            // (8, N)
{
    __shared__ __align__(16) _Float16 lds_h1[KNB * H1S];   // 18 KB  h1 rows (f16)
    __shared__ __align__(16) _Float16 lds_w2t[HID * W2S];  // 9 KB   W2^T [n][k] (f16)
    __shared__ float lds_rep[KNB * 4];
    __shared__ float lds_w1[4 * HID];
    __shared__ float lds_b1[HID];
    __shared__ float lds_w3[HID];
    __shared__ float lds_kv[KNB];
    __shared__ float lds_msk[KNB];
    __shared__ int   lds_idx[KNB];
    __shared__ float lds_p[8 * KNB];
    __shared__ float lds_acc[9];

    const int n    = blockIdx.x;
    const int tid  = threadIdx.x;
    const int lane = tid & 31;
    const int wv   = tid >> 5;

    // ---------------- stage 0: stage weights + gather neighbor features -------
    lds_w1[tid] = W1[tid];                       // 256 == blockDim, exact fit
    if (tid < HID)              lds_b1[tid]       = b1[tid];
    else if (tid < 2 * HID)     lds_w3[tid - HID] = W3[tid - HID];

    // W2 transposed into LDS as f16: lds_w2t[nout*W2S + kin] = W2[kin*64 + nout]
    for (int i = tid; i < HID * HID; i += 256) {
        int no = i >> 6, ki = i & 63;
        lds_w2t[no * W2S + ki] = (_Float16)W2[ki * HID + no];
    }

    const float cx = coords[n];
    const float cy = coords[N_PTS + n];
    if (tid < KNB) {
        int   id = nbr_idx[(size_t)n * KNB + tid];
        float m  = nbr_msk[(size_t)n * KNB + tid] ? 1.0f : 0.0f;
        lds_idx[tid]         = id;
        lds_msk[tid]         = m;
        lds_rep[tid * 4 + 0] = cx;
        lds_rep[tid * 4 + 1] = cy;
        lds_rep[tid * 4 + 2] = coords[id];
        lds_rep[tid * 4 + 3] = coords[N_PTS + id];
    }
    __syncthreads();

    // ---------------- stage 1: h1 = gelu(rep @ W1 + b1)  -> f16 LDS -----------
    {
        int   r  = tid >> 1;
        int   c0 = (tid & 1) * 32;
        float f0 = lds_rep[r * 4 + 0], f1 = lds_rep[r * 4 + 1];
        float f2 = lds_rep[r * 4 + 2], f3 = lds_rep[r * 4 + 3];
        #pragma unroll 8
        for (int c = c0; c < c0 + 32; ++c) {
            float s = lds_b1[c] + f0 * lds_w1[0 * HID + c] + f1 * lds_w1[1 * HID + c]
                                + f2 * lds_w1[2 * HID + c] + f3 * lds_w1[3 * HID + c];
            lds_h1[r * H1S + c] = (_Float16)gelu_tanh(s);
        }
    }
    __syncthreads();

    // ---------------- stage 2: h2 = gelu(h1 @ W2 + b2) via WMMA ---------------
    // Wave wv owns rows [16*wv, 16*wv+16). 4 N-tiles x 2 K-steps = 8 WMMA/wave.
    // A layout (16-bit 16x32): lane = M (mod 16); elements 0..7 -> K = ks*32 + khi + e,
    // elements 8..15 -> +16; khi = 8 for lanes 16..31.
    v16h afrag[2];
    {
        int row = wv * 16 + (lane & 15);
        int khi = (lane >= 16) ? 8 : 0;
        #pragma unroll
        for (int ks = 0; ks < 2; ++ks) {
            const _Float16* p0 = &lds_h1[row * H1S + ks * 32 + khi];
            v8h lo = *(const v8h*)p0;
            v8h hi = *(const v8h*)(p0 + 16);
            v16h a;
            #pragma unroll
            for (int i = 0; i < 8; ++i) { a[i] = lo[i]; a[i + 8] = hi[i]; }
            afrag[ks] = a;
        }
    }

    float kpart[8];
    #pragma unroll
    for (int v = 0; v < 8; ++v) kpart[v] = 0.0f;

    const int colLane = lane & 15;
    const int kbaseB  = (lane < 16) ? 0 : 8;    // B layout: K = khi+e, +16 for e>=8 (matches A)
    #pragma unroll
    for (int nt = 0; nt < 4; ++nt) {
        int   col  = nt * 16 + colLane;
        float bias = b2[col];
        v8f acc;
        #pragma unroll
        for (int i = 0; i < 8; ++i) acc[i] = bias;   // C layout: whole frag shares N=col
        #pragma unroll
        for (int ks = 0; ks < 2; ++ks) {
            const _Float16* pb = &lds_w2t[col * W2S + ks * 32 + kbaseB];
            v8h lo = *(const v8h*)pb;
            v8h hi = *(const v8h*)(pb + 16);
            v16h bfrag;
            #pragma unroll
            for (int i = 0; i < 8; ++i) { bfrag[i] = lo[i]; bfrag[i + 8] = hi[i]; }
            acc = __builtin_amdgcn_wmma_f32_16x16x32_f16(
                      /*neg_a=*/false, afrag[ks], /*neg_b=*/false, bfrag,
                      /*c_mod=*/(short)0, acc, /*reuse_a=*/false, /*reuse_b=*/false);
        }
        // fused: gelu on accumulator + W3 dot partial (this lane's column)
        float w3c = lds_w3[col];
        #pragma unroll
        for (int v = 0; v < 8; ++v) kpart[v] += gelu_tanh(acc[v]) * w3c;
    }

    // reduce kpart across the 16 lanes that share each row-half (xor stays in half)
    #pragma unroll
    for (int v = 0; v < 8; ++v) {
        float s = kpart[v];
        s += __shfl_xor(s, 1, 32);
        s += __shfl_xor(s, 2, 32);
        s += __shfl_xor(s, 4, 32);
        s += __shfl_xor(s, 8, 32);
        kpart[v] = s;
    }
    const float b3v = b3[0];
    if (colLane == 0) {
        // lanes 0 / 16 hold rows wv*16 + v and wv*16 + 8 + v  (C layout M mapping)
        int rbase = wv * 16 + ((lane >= 16) ? 8 : 0);
        #pragma unroll
        for (int v = 0; v < 8; ++v) lds_kv[rbase + v] = kpart[v] + b3v;
    }
    __syncthreads();

    // ---------------- stage 3: masked einsum over neighbors -------------------
    if (tid < KNB) {
        float f  = lds_kv[tid] * lds_msk[tid];
        int   id = lds_idx[tid];
        #pragma unroll
        for (int b = 0; b < 8; ++b)
            lds_p[b * KNB + tid] = f * samples[b * N_PTS + id];
    }
    __syncthreads();
    {
        // wave wv reduces batch row wv across 128 slots
        float s = lds_p[wv * KNB + lane]      + lds_p[wv * KNB + lane + 32]
                + lds_p[wv * KNB + lane + 64] + lds_p[wv * KNB + lane + 96];
        #pragma unroll
        for (int off = 16; off > 0; off >>= 1) s += __shfl_xor(s, off, 32);
        if (lane == 0) lds_acc[wv] = s;
        if (wv == 0) {
            float c = lds_msk[lane] + lds_msk[lane + 32]
                    + lds_msk[lane + 64] + lds_msk[lane + 96];
            #pragma unroll
            for (int off = 16; off > 0; off >>= 1) c += __shfl_xor(c, off, 32);
            if (lane == 0) lds_acc[8] = c;
        }
    }
    __syncthreads();

    if (tid < 8) {
        float cnt = fmaxf(lds_acc[8], 1.0f);
        out[tid * N_PTS + n] = lds_acc[tid] / cnt;
    }
}

extern "C" void kernel_launch(void* const* d_in, const int* in_sizes, int n_in,
                              void* d_out, int out_size, void* d_ws, size_t ws_size,
                              hipStream_t stream) {
    const float*   coords  = (const float*)d_in[0];
    const float*   samples = (const float*)d_in[1];
    /* d_in[2] = sigma: unused by the reference computation */
    const float*   W1      = (const float*)d_in[3];
    const float*   b1      = (const float*)d_in[4];
    const float*   W2      = (const float*)d_in[5];
    const float*   b2      = (const float*)d_in[6];
    const float*   W3      = (const float*)d_in[7];
    const float*   b3      = (const float*)d_in[8];
    const int*     nbr     = (const int*)d_in[9];
    const uint8_t* msk     = (const uint8_t*)d_in[10];
    float*         out     = (float*)d_out;

    mpscore_kernel<<<N_PTS, 256, 0, stream>>>(coords, samples, W1, b1, W2, b2,
                                              W3, b3, nbr, msk, out);
}
